// Feature_Grid_Model_19559281066529
// MI455X (gfx1250) — matrix operand order repaired and checked by reading, not compiled, not run on hardware
//
#include <hip/hip_runtime.h>

typedef __attribute__((ext_vector_type(16))) _Float16 v16h;
typedef __attribute__((ext_vector_type(8)))  _Float16 v8h;
typedef __attribute__((ext_vector_type(8)))  float    v8f;
typedef __attribute__((ext_vector_type(4)))  unsigned u32x4;
typedef __attribute__((ext_vector_type(8)))  int      i32x8;
typedef __attribute__((ext_vector_type(4)))  int      i32x4;

#define NPTS   1000000
#define GRID3  (64*64*64)
#define INCH   79
#define XPAD   104   // activation row stride (f16): 52 dwords -> conflict-free b128 loads
#define WHPAD  40    // hidden weight row stride (f16): 20 dwords -> conflict-free
#define WAVES  6
#define BLK    (WAVES*32)

// weight blob layout (bytes) -- identical in global (d_ws) and LDS
#define W0_BYTES   (32*XPAD*2)        // 6656
#define WH_OFF     W0_BYTES
#define WH_BYTES   (3*32*WHPAD*2)     // 7680
#define B0_OFF     (WH_OFF + WH_BYTES)      // 14336
#define BH_OFF     (B0_OFF + 32*4)          // 14464
#define BLOB_BYTES (BH_OFF + 96*4)          // 14848
#define GRID_BYTES (GRID3*16*2)             // 8 MB

union AFrag { v16h v; u32x4 q[2]; };

// ---------------------------------------------------------------------------
// Prep 1: (C=16, 64,64,64) f32 -> (z,y,x, C=16) f16. One corner's 16 channels
// become one contiguous 32B read; source reads stay coalesced per channel.
// ---------------------------------------------------------------------------
__global__ void reorder_grid_kernel(const float* __restrict__ g,
                                    _Float16* __restrict__ o) {
  int idx = blockIdx.x * blockDim.x + threadIdx.x;
  if (idx >= GRID3) return;
  union { _Float16 h[8]; u32x4 q; } t0, t1;
#pragma unroll
  for (int c = 0; c < 8; ++c) t0.h[c] = (_Float16)g[(size_t)c * GRID3 + idx];
#pragma unroll
  for (int c = 0; c < 8; ++c) t1.h[c] = (_Float16)g[(size_t)(c + 8) * GRID3 + idx];
  u32x4* op = (u32x4*)(o + (size_t)idx * 16);
  op[0] = t0.q;
  op[1] = t1.q;
}

// ---------------------------------------------------------------------------
// Prep 2 (single block): build the LDS-image weight blob in d_ws.
// Input-channel permutation (storage col c -> reference k):
//   c<16: 63+c (feats) | 16..18: c-16 (coords) | 20..79: c-17 (PE) | else 0
// ---------------------------------------------------------------------------
__global__ void prep_weights_kernel(const float* __restrict__ W0,
                                    const float* __restrict__ b0,
                                    const float* __restrict__ Wh,
                                    const float* __restrict__ bh,
                                    unsigned char* __restrict__ blob) {
  const int tid = threadIdx.x;
  _Float16* w0 = (_Float16*)blob;
  for (int i = tid; i < 32 * XPAD; i += 256) {
    int n = i / XPAD, c = i % XPAD;
    int k; bool valid = true;
    if (c < 16)       k = 63 + c;
    else if (c < 19)  k = c - 16;
    else if (c == 19) { k = 0; valid = false; }
    else if (c < 80)  k = c - 17;
    else              { k = 0; valid = false; }
    w0[i] = valid ? (_Float16)W0[k * 32 + n] : (_Float16)0.0f;
  }
  _Float16* wh = (_Float16*)(blob + WH_OFF);
  for (int i = tid; i < 3 * 32 * WHPAD; i += 256) {
    int l = i / (32 * WHPAD), r = i % (32 * WHPAD), n = r / WHPAD, k = r % WHPAD;
    wh[i] = (k < 32) ? (_Float16)Wh[(l * 32 + k) * 32 + n] : (_Float16)0.0f;
  }
  float* bb0 = (float*)(blob + B0_OFF);
  if (tid < 32) bb0[tid] = b0[tid];
  float* bbh = (float*)(blob + BH_OFF);
  if (tid < 96) bbh[tid] = bh[tid];
}

// pack two f32 -> packed f16 pair (one dword)
__device__ __forceinline__ unsigned pk2(float a, float b) {
  union { _Float16 h[2]; unsigned u; } t;
  t.h[0] = (_Float16)a; t.h[1] = (_Float16)b;
  return t.u;
}

// snake on a D-fragment (8 VGPRs), store f16 row-major into LDS (stride XPAD)
__device__ __forceinline__ void snake_store(v8f a, _Float16* base) {
#pragma unroll
  for (int r = 0; r < 8; ++r) {
    float s = __sinf(a[r]);
    base[r * XPAD] = (_Float16)(0.5f * a[r] + s * s);
  }
}

// ---------------------------------------------------------------------------
// Main fused kernel: 6 waves/block, 32 points/wave, WMMA f16->f32 MLP.
// Weights arrive in LDS via one Tensor Data Mover DMA (TENSORcnt).
// ---------------------------------------------------------------------------
__global__ __launch_bounds__(BLK, 2) void feature_mlp_kernel(
    const float* __restrict__ coords, const _Float16* __restrict__ grid,
    const unsigned char* __restrict__ wblob,
    const float* __restrict__ Wf, const float* __restrict__ bfin,
    float* __restrict__ out) {
  __shared__ __align__(16) unsigned char sWblob[BLOB_BYTES];
  __shared__ __align__(16) _Float16 sX[WAVES][32][XPAD];

  const int tid = threadIdx.x;

  // ---- stage full weight blob into LDS via the Tensor Data Mover ----
#if defined(__has_builtin) && __has_builtin(__builtin_amdgcn_tensor_load_to_lds)
  if (tid < 32) {  // one wave issues the DMA (EXEC ignored by TDM)
    unsigned lds_base = (unsigned)(unsigned long long)(size_t)&sWblob[0];
    unsigned long long ga = (unsigned long long)(size_t)wblob;
    const unsigned NE = BLOB_BYTES / 8;  // 8-byte elements
    u32x4 g0;
    g0[0] = 1u;                                       // count=1 (user D#)
    g0[1] = lds_base;                                 // lds_addr
    g0[2] = (unsigned)(ga & 0xFFFFFFFFull);           // global_addr[31:0]
    g0[3] = (unsigned)((ga >> 32) & 0x01FFFFFFull) | (2u << 30);  // addr hi | type=2
    i32x8 g1;
    g1[0] = (int)(3u << 16);                          // data_size = 8B
    g1[1] = (int)((NE & 0xFFFFu) << 16);              // tensor_dim0 lo
    g1[2] = (int)(((NE >> 16) & 0xFFFFu) | (1u << 16)); // dim0 hi | tensor_dim1=1
    g1[3] = (int)((NE & 0xFFFFu) << 16);              // tile_dim0 = NE
    g1[4] = 1;                                        // tile_dim1=1, tile_dim2=0
    g1[5] = (int)NE;                                  // tensor_dim0_stride lo32
    g1[6] = (int)((NE & 0xFFFFu) << 16);              // stride hi | dim1_stride lo
    g1[7] = 0;
    i32x4 z4 = {0, 0, 0, 0};
#if __clang_major__ >= 23
    i32x8 z8 = {0, 0, 0, 0, 0, 0, 0, 0};
    __builtin_amdgcn_tensor_load_to_lds(g0, g1, z4, z4, z8, 0);
#else
    __builtin_amdgcn_tensor_load_to_lds(g0, g1, z4, z4, 0);
#endif
    __builtin_amdgcn_s_wait_tensorcnt(0);
  }
#else
  for (int i = tid; i < BLOB_BYTES / 16; i += BLK)
    ((u32x4*)sWblob)[i] = ((const u32x4*)wblob)[i];
#endif
  __syncthreads();

  const _Float16 (*sW0)[XPAD] = (const _Float16(*)[XPAD])(sWblob);
  const _Float16 (*sWh)[32][WHPAD] = (const _Float16(*)[32][WHPAD])(sWblob + WH_OFF);
  const float* sB0 = (const float*)(sWblob + B0_OFF);
  const float* sBh = (const float*)(sWblob + BH_OFF);  // [l*32+n]

  const int wave = tid >> 5;
  const int lane = tid & 31;
  const int hf   = lane >> 4;   // lane half (K/N split in WMMA layouts)
  const int nn   = lane & 15;
  const long long p = (long long)blockIdx.x * BLK + tid;
  const bool live = p < NPTS;

  float cx = live ? coords[p * 3 + 0] : 0.f;
  float cy = live ? coords[p * 3 + 1] : 0.f;
  float cz = live ? coords[p * 3 + 2] : 0.f;

  // ---- trilinear grid sample from channel-last f16 grid ----
  float feats[16];
#pragma unroll
  for (int c = 0; c < 16; ++c) feats[c] = 0.f;
  {
    float xf = (cx + 1.f) * 32.f - 0.5f;
    float yf = (cy + 1.f) * 32.f - 0.5f;
    float zf = (cz + 1.f) * 32.f - 0.5f;
    float x0 = floorf(xf), y0 = floorf(yf), z0 = floorf(zf);
    float wx = xf - x0, wy = yf - y0, wz = zf - z0;
    int ix = (int)x0, iy = (int)y0, iz = (int)z0;
#pragma unroll
    for (int corner = 0; corner < 8; ++corner) {
      int dx = corner & 1, dy = (corner >> 1) & 1, dz = corner >> 2;
      int xi = ix + dx, yi = iy + dy, zi = iz + dz;
      bool ok = (xi >= 0) & (xi < 64) & (yi >= 0) & (yi < 64) & (zi >= 0) & (zi < 64);
      int xc = min(max(xi, 0), 63), yc = min(max(yi, 0), 63), zc = min(max(zi, 0), 63);
      float w = (dx ? wx : 1.f - wx) * (dy ? wy : 1.f - wy) * (dz ? wz : 1.f - wz);
      w = ok ? w : 0.f;
      const v8h* gp = (const v8h*)(grid + (((size_t)zc * 64 + yc) * 64 + xc) * 16);
      v8h f0 = gp[0], f1 = gp[1];
#pragma unroll
      for (int c = 0; c < 8; ++c) feats[c]     += w * (float)f0[c];
#pragma unroll
      for (int c = 0; c < 8; ++c) feats[c + 8] += w * (float)f1[c];
    }
  }

  // ---- build permuted input row in LDS with packed stores ----
  _Float16* xr = &sX[wave][lane][0];
  {
    union { _Float16 h[8]; u32x4 q; } f0, f1;
#pragma unroll
    for (int c = 0; c < 8; ++c) f0.h[c] = (_Float16)feats[c];
#pragma unroll
    for (int c = 0; c < 8; ++c) f1.h[c] = (_Float16)feats[c + 8];
    *(u32x4*)(xr + 0) = f0.q;                       // cols 0..15 : feats
    *(u32x4*)(xr + 8) = f1.q;
    *(unsigned*)(xr + 16) = pk2(cx, cy);            // cols 16..19: coords, pad
    *(unsigned*)(xr + 18) = pk2(cz, 0.f);
    float fr = 1.f;
#pragma unroll
    for (int l = 0; l < 10; ++l) {                  // cols 20..79: PE
      float ax = cx * fr, ay = cy * fr, az = cz * fr;
      float sx = __sinf(ax), sy = __sinf(ay), sz = __sinf(az);
      float ox = __cosf(ax), oy = __cosf(ay), oz = __cosf(az);
      *(unsigned*)(xr + 20 + l * 6)     = pk2(sx, sy);
      *(unsigned*)(xr + 20 + l * 6 + 2) = pk2(sz, ox);
      *(unsigned*)(xr + 20 + l * 6 + 4) = pk2(oy, oz);
      fr *= 2.f;
    }
    u32x4 z4 = {0u, 0u, 0u, 0u};                    // cols 80..95: zero pad
    *(u32x4*)(xr + 80) = z4;
    *(u32x4*)(xr + 88) = z4;
  }

  // wave-local LDS turnaround (DS ops are in-order per wave)
  asm volatile("s_wait_dscnt 0" ::: "memory");

  v8f acc00, acc01, acc10, acc11;

  // ---- layer 0: (32pts x 96) @ (96 x 32), 3 K-chunks x 4 tiles ----
  {
    float bv0 = sB0[nn], bv1 = sB0[16 + nn];
#pragma unroll
    for (int i = 0; i < 8; ++i) { acc00[i] = bv0; acc01[i] = bv1; acc10[i] = bv0; acc11[i] = bv1; }
#pragma unroll
    for (int kc = 0; kc < 3; ++kc) {
      AFrag a0, a1, fb0, fb1;
      const _Float16* r0 = &sX[wave][nn][kc * 32];
      const _Float16* r1 = &sX[wave][16 + nn][kc * 32];
      a0.q[0] = *(const u32x4*)(r0 + hf * 8);  a0.q[1] = *(const u32x4*)(r0 + 16 + hf * 8);
      a1.q[0] = *(const u32x4*)(r1 + hf * 8);  a1.q[1] = *(const u32x4*)(r1 + 16 + hf * 8);
      const _Float16* w0r = &sW0[nn][kc * 32 + hf * 8];
      const _Float16* w1r = &sW0[16 + nn][kc * 32 + hf * 8];
      fb0.q[0] = *(const u32x4*)(w0r); fb0.q[1] = *(const u32x4*)(w0r + 16);
      fb1.q[0] = *(const u32x4*)(w1r); fb1.q[1] = *(const u32x4*)(w1r + 16);
      acc00 = __builtin_amdgcn_wmma_f32_16x16x32_f16(false, a0.v, false, fb0.v, (short)0, acc00, false, false);
      acc01 = __builtin_amdgcn_wmma_f32_16x16x32_f16(false, a0.v, false, fb1.v, (short)0, acc01, false, false);
      acc10 = __builtin_amdgcn_wmma_f32_16x16x32_f16(false, a1.v, false, fb0.v, (short)0, acc10, false, false);
      acc11 = __builtin_amdgcn_wmma_f32_16x16x32_f16(false, a1.v, false, fb1.v, (short)0, acc11, false, false);
    }
    snake_store(acc00, &sX[wave][0  + hf * 8][nn]);
    snake_store(acc01, &sX[wave][0  + hf * 8][16 + nn]);
    snake_store(acc10, &sX[wave][16 + hf * 8][nn]);
    snake_store(acc11, &sX[wave][16 + hf * 8][16 + nn]);
  }

  // ---- hidden layers: (32 x 32) @ (32 x 32), one WMMA per tile ----
#pragma unroll
  for (int layer = 0; layer < 3; ++layer) {
    asm volatile("s_wait_dscnt 0" ::: "memory");
    AFrag a0, a1, fb0, fb1;
    const _Float16* r0 = &sX[wave][nn][0];
    const _Float16* r1 = &sX[wave][16 + nn][0];
    a0.q[0] = *(const u32x4*)(r0 + hf * 8);  a0.q[1] = *(const u32x4*)(r0 + 16 + hf * 8);
    a1.q[0] = *(const u32x4*)(r1 + hf * 8);  a1.q[1] = *(const u32x4*)(r1 + 16 + hf * 8);
    const _Float16* w0r = &sWh[layer][nn][hf * 8];
    const _Float16* w1r = &sWh[layer][16 + nn][hf * 8];
    fb0.q[0] = *(const u32x4*)(w0r); fb0.q[1] = *(const u32x4*)(w0r + 16);
    fb1.q[0] = *(const u32x4*)(w1r); fb1.q[1] = *(const u32x4*)(w1r + 16);
    float bv0 = sBh[layer * 32 + nn], bv1 = sBh[layer * 32 + 16 + nn];
#pragma unroll
    for (int i = 0; i < 8; ++i) { acc00[i] = bv0; acc01[i] = bv1; acc10[i] = bv0; acc11[i] = bv1; }
    acc00 = __builtin_amdgcn_wmma_f32_16x16x32_f16(false, a0.v, false, fb0.v, (short)0, acc00, false, false);
    acc01 = __builtin_amdgcn_wmma_f32_16x16x32_f16(false, a0.v, false, fb1.v, (short)0, acc01, false, false);
    acc10 = __builtin_amdgcn_wmma_f32_16x16x32_f16(false, a1.v, false, fb0.v, (short)0, acc10, false, false);
    acc11 = __builtin_amdgcn_wmma_f32_16x16x32_f16(false, a1.v, false, fb1.v, (short)0, acc11, false, false);
    snake_store(acc00, &sX[wave][0  + hf * 8][nn]);
    snake_store(acc01, &sX[wave][0  + hf * 8][16 + nn]);
    snake_store(acc10, &sX[wave][16 + hf * 8][nn]);
    snake_store(acc11, &sX[wave][16 + hf * 8][16 + nn]);
  }

  // ---- final layer: per-lane dot(x[32], Wf) + bf (Wf via scalar loads) ----
  asm volatile("s_wait_dscnt 0" ::: "memory");
  float accf = bfin[0];
  const _Float16* mrow = &sX[wave][lane][0];
#pragma unroll
  for (int k = 0; k < 32; ++k) accf += (float)mrow[k] * Wf[k];
  if (live) out[p] = accf;
}

// ---------------------------------------------------------------------------
extern "C" void kernel_launch(void* const* d_in, const int* in_sizes, int n_in,
                              void* d_out, int out_size, void* d_ws, size_t ws_size,
                              hipStream_t stream) {
  const float* coords = (const float*)d_in[0];
  const float* fgrid  = (const float*)d_in[1];
  const float* W0     = (const float*)d_in[2];
  const float* b0     = (const float*)d_in[3];
  const float* Wh     = (const float*)d_in[4];
  const float* bh     = (const float*)d_in[5];
  const float* Wf     = (const float*)d_in[6];
  const float* bfp    = (const float*)d_in[7];
  float* out = (float*)d_out;
  _Float16* gridf16   = (_Float16*)d_ws;                        // 8 MB
  unsigned char* blob = (unsigned char*)d_ws + GRID_BYTES;      // 14848 B

  reorder_grid_kernel<<<(GRID3 + 255) / 256, 256, 0, stream>>>(fgrid, gridf16);
  prep_weights_kernel<<<1, 256, 0, stream>>>(W0, b0, Wh, bh, blob);

  int nblocks = (NPTS + BLK - 1) / BLK;
  feature_mlp_kernel<<<nblocks, BLK, 0, stream>>>(coords, gridf16, blob, Wf,
                                                  bfp, out);
}
